// ShIeLD_25082609008858
// MI455X (gfx1250) — matrix operand order repaired and checked by reading, not compiled, not run on hardware
//
#include <hip/hip_runtime.h>
#include <hip/hip_bf16.h>

// GATv2Conv forward for MI455X (gfx1250, wave32).
// GEMMs use v_wmma_f32_16x16x32_bf16; edge phase is wave-per-edge with
// lane==channel coalesced gathers and f32 atomics (working set lives in 192MB L2).

typedef __attribute__((ext_vector_type(16))) __bf16          v16bf;
typedef __attribute__((ext_vector_type(16))) unsigned short  v16u;
typedef __attribute__((ext_vector_type(8)))  float           v8f;

#define C_CH   64
#define F_IN   256

__device__ __forceinline__ unsigned short f2bf(float f) {
    unsigned u = __float_as_uint(f);
    u += 0x7FFFu + ((u >> 16) & 1u);     // round-to-nearest-even
    return (unsigned short)(u >> 16);
}

__device__ __forceinline__ void atomicMaxF(float* addr, float val) {
    unsigned* ua = (unsigned*)addr;
    unsigned old = *ua;
    while (__uint_as_float(old) < val) {
        unsigned assumed = old;
        old = atomicCAS(ua, assumed, __float_as_uint(val));
        if (old == assumed) break;
    }
}

// ---------------------------------------------------------------- fills
__global__ void fill_f(float* __restrict__ p, size_t n, float v) {
    size_t i  = (size_t)blockIdx.x * blockDim.x + threadIdx.x;
    size_t st = (size_t)gridDim.x * blockDim.x;
    for (; i < n; i += st) p[i] = v;
}

// ------------------------------------------------- pre-pack W into B-fragment layout
// B fragment (32x16 bf16, K x N): lane holds column N = lane&15, vgpr v holds
// K = (v>=4?16:0) + 8*(lane>>4) + (v&3)*2 (+0,+1).  Frag id = mat*32 + kkIdx*4 + tile. 16 bf16 per lane.
__global__ void pack_w(const float* __restrict__ Wl, const float* __restrict__ Wr,
                       unsigned short* __restrict__ packed) {
    int t = blockIdx.x * blockDim.x + threadIdx.x;     // 0..2047
    if (t >= 2 * 8 * 4 * 32) return;
    int lane  = t & 31;
    int tile  = (t >> 5) & 3;
    int kkIdx = (t >> 7) & 7;
    int mat   = t >> 10;
    const float* W = mat ? Wr : Wl;
    unsigned short* dst = packed + ((size_t)(mat * 32 + kkIdx * 4 + tile) * 32 + lane) * 16;
    int n = tile * 16 + (lane & 15);                   // B[k][n] = W[n][k]
    #pragma unroll
    for (int v = 0; v < 8; v++) {
        int kB = kkIdx * 32 + ((v >= 4) ? 16 : 0) + 8 * (lane >> 4) + (v & 3) * 2;
        dst[2 * v]     = f2bf(W[(size_t)n * F_IN + kB]);
        dst[2 * v + 1] = f2bf(W[(size_t)n * F_IN + kB + 1]);
    }
}

// ------------------------------------------------- fused xl/xr GEMM (bf16 WMMA)
// One wave -> rows [m0,m0+16), all 64 channels of BOTH xl and xr (x read once).
__global__ void gemm_xlxr(const float* __restrict__ x,
                          const unsigned short* __restrict__ packed,
                          const float* __restrict__ bl, const float* __restrict__ br,
                          float* __restrict__ xl, float* __restrict__ xr, int Nn) {
    const int lane = threadIdx.x & 31;
    const int wave = threadIdx.x >> 5;                 // 0..7
    const int m0   = (blockIdx.x * 8 + wave) * 16;
    if (m0 >= Nn) return;
    const int half = lane >> 4;
    const int r    = lane & 15;
    const int row  = min(m0 + r, Nn - 1);
    const float* xrow = x + (size_t)row * F_IN;
    const v16u* pB = (const v16u*)packed;

    v8f accl[4] = {};
    v8f accr[4] = {};

    for (int kk = 0; kk < F_IN; kk += 32) {
        // A fragment: 16x32 bf16 layout — lane (half,r): vgpr j holds
        // K = kk + (j>=4?16:0) + half*8 + (j&3)*2  (two packed bf16)
        v16u au;
        #pragma unroll
        for (int j = 0; j < 8; j++) {
            int k = kk + ((j >= 4) ? 16 : 0) + half * 8 + (j & 3) * 2;
            au[2 * j]     = f2bf(xrow[k]);
            au[2 * j + 1] = f2bf(xrow[k + 1]);
        }
        v16bf a = __builtin_bit_cast(v16bf, au);
        int kkIdx = kk >> 5;
        #pragma unroll
        for (int t = 0; t < 4; t++) {
            v16bf bL = __builtin_bit_cast(v16bf, pB[(size_t)(      kkIdx * 4 + t) * 32 + lane]);
            v16bf bR = __builtin_bit_cast(v16bf, pB[(size_t)(32 +  kkIdx * 4 + t) * 32 + lane]);
            accl[t] = __builtin_amdgcn_wmma_f32_16x16x32_bf16(false, a, false, bL,
                                                              (short)0, accl[t], false, false);
            accr[t] = __builtin_amdgcn_wmma_f32_16x16x32_bf16(false, a, false, bR,
                                                              (short)0, accr[t], false, false);
        }
    }
    // C/D layout: vgpr i, lanes 0-15 -> M=i, lanes 16-31 -> M=i+8; N = lane&15
    #pragma unroll
    for (int t = 0; t < 4; t++) {
        int col = t * 16 + r;
        float bL = bl[col], bR = br[col];
        #pragma unroll
        for (int i = 0; i < 8; i++) {
            int rw = m0 + i + 8 * half;
            if (rw < Nn) {
                xl[(size_t)rw * C_CH + col] = accl[t][i] + bL;
                xr[(size_t)rw * C_CH + col] = accr[t][i] + bR;
            }
        }
    }
}

// ------------------------------------------------- self-loop attr stats
__global__ void edge_stats(const int* __restrict__ dst, const float* __restrict__ eattr,
                           float* __restrict__ cnt, float* __restrict__ asum, int E) {
    int e = blockIdx.x * blockDim.x + threadIdx.x;
    if (e < E) {
        int d = dst[e];
        atomicAdd(&cnt[d], 1.0f);
        atomicAdd(&asum[d], eattr[e]);
    }
}

__global__ void loopattr_k(float* __restrict__ asum, const float* __restrict__ cnt, int Nn) {
    int n = blockIdx.x * blockDim.x + threadIdx.x;
    if (n < Nn) asum[n] = asum[n] / fmaxf(cnt[n], 1.0f);
}

// ------------------------------------------------- attention score + segment max
__global__ void edge_score(const int* __restrict__ src, const int* __restrict__ dst,
                           const float* __restrict__ eattr, const float* __restrict__ loop_attr,
                           const float* __restrict__ xl, const float* __restrict__ xr,
                           const float* __restrict__ We, const float* __restrict__ att,
                           float* __restrict__ escore, float* __restrict__ emax, int E, int Nn) {
    int w    = (blockIdx.x * blockDim.x + threadIdx.x) >> 5;
    int lane = threadIdx.x & 31;
    if (w >= E + Nn) return;
    int s, d; float a;
    if (w < E) { s = src[w]; d = dst[w]; a = eattr[w]; }
    else       { s = d = w - E;          a = loop_attr[s]; }
    float sc = 0.0f;
    #pragma unroll
    for (int h = 0; h < 2; h++) {
        int c = lane + 32 * h;
        float m = xl[(size_t)s * C_CH + c] + xr[(size_t)d * C_CH + c] + We[c] * a;
        m = (m > 0.0f) ? m : 0.2f * m;     // leaky_relu(0.2)
        sc += att[c] * m;
    }
    #pragma unroll
    for (int off = 16; off > 0; off >>= 1) sc += __shfl_xor(sc, off, 32);
    if (lane == 0) { escore[w] = sc; atomicMaxF(&emax[d], sc); }
}

// ------------------------------------------------- exp + segment denom (ee in place)
__global__ void expdenom_k(const int* __restrict__ dst, float* __restrict__ escore,
                           const float* __restrict__ emax, float* __restrict__ denom,
                           int E, int Nn) {
    int e = blockIdx.x * blockDim.x + threadIdx.x;
    if (e >= E + Nn) return;
    int d = (e < E) ? dst[e] : (e - E);
    float v = __expf(escore[e] - emax[d]);
    escore[e] = v;
    atomicAdd(&denom[d], v);
}

// ------------------------------------------------- alpha * xl[src] scatter-add
__global__ void scatter_k(const int* __restrict__ src, const int* __restrict__ dst,
                          const float* __restrict__ ee, const float* __restrict__ denom,
                          const float* __restrict__ xl, float* __restrict__ agg,
                          int E, int Nn) {
    int w    = (blockIdx.x * blockDim.x + threadIdx.x) >> 5;
    int lane = threadIdx.x & 31;
    if (w >= E + Nn) return;
    int s, d;
    if (w < E) { s = src[w]; d = dst[w]; }
    else       { s = d = w - E; }
    float al = ee[w] / denom[d];
    atomicAdd(&agg[(size_t)d * C_CH + lane],      al * xl[(size_t)s * C_CH + lane]);
    atomicAdd(&agg[(size_t)d * C_CH + lane + 32], al * xl[(size_t)s * C_CH + lane + 32]);
}

// ------------------------------------------------- relu + mean pool (per channel)
// Launch with (grid*block) % 64 == 0 so each thread's channel is fixed.
__global__ void pool_k(const float* __restrict__ agg, const float* __restrict__ b_conv,
                       float* __restrict__ pooled, size_t total) {
    size_t stride = (size_t)gridDim.x * blockDim.x;
    size_t i = (size_t)blockIdx.x * blockDim.x + threadIdx.x;
    int c = (int)(i & 63);
    float bc = b_conv[c];
    float local = 0.0f;
    for (; i < total; i += stride) local += fmaxf(agg[i] + bc, 0.0f);
    atomicAdd(&pooled[c], local);
}

// ------------------------------------------------- head: mean, linear, softmax
__global__ void head_k(const float* __restrict__ pooled, const float* __restrict__ W_head,
                       const float* __restrict__ b_head, float* __restrict__ out, float invN) {
    if (threadIdx.x == 0 && blockIdx.x == 0) {
        float l0 = b_head[0], l1 = b_head[1];
        for (int c = 0; c < C_CH; c++) {
            float m = pooled[c] * invN;
            l0 += W_head[c] * m;
            l1 += W_head[C_CH + c] * m;
        }
        float mx = fmaxf(l0, l1);
        float e0 = __expf(l0 - mx), e1 = __expf(l1 - mx);
        float s = e0 + e1;
        out[0] = e0 / s;
        out[1] = e1 / s;
    }
}

extern "C" void kernel_launch(void* const* d_in, const int* in_sizes, int n_in,
                              void* d_out, int out_size, void* d_ws, size_t ws_size,
                              hipStream_t stream) {
    const float* x      = (const float*)d_in[0];
    const int*   ei     = (const int*)d_in[1];
    const float* eattr  = (const float*)d_in[2];
    const float* Wl     = (const float*)d_in[3];
    const float* bl     = (const float*)d_in[4];
    const float* Wr     = (const float*)d_in[5];
    const float* br     = (const float*)d_in[6];
    const float* We     = (const float*)d_in[7];
    const float* att    = (const float*)d_in[8];
    const float* b_conv = (const float*)d_in[9];
    const float* W_head = (const float*)d_in[10];
    const float* b_head = (const float*)d_in[11];

    const int N  = in_sizes[0] / F_IN;     // 50000
    const int E  = in_sizes[2];            // 800000
    const int ET = E + N;
    const int* src = ei;
    const int* dst = ei + E;

    // -------- workspace layout (floats) --------
    float* W = (float*)d_ws;
    size_t o = 0;
    float* xlp    = W + o; o += (size_t)N * C_CH;
    float* xrp    = W + o; o += (size_t)N * C_CH;
    float* agg    = W + o; o += (size_t)N * C_CH;   // zero region starts at agg
    float* cnt    = W + o; o += N;
    float* asum   = W + o; o += N;                  // becomes loop_attr in place
    float* denom  = W + o; o += N;
    float* pooled = W + o; o += 64;
    size_t zeroCount = (size_t)N * C_CH + 3 * (size_t)N + 64;
    float* emax   = W + o; o += N;
    float* escore = W + o; o += ET;                 // becomes ee in place
    unsigned short* packed = (unsigned short*)(W + o); // 2*64*256 bf16 = 16384 floats
    o += 16384;

    // -------- pipeline (stream-ordered) --------
    pack_w<<<8, 256, 0, stream>>>(Wl, Wr, packed);
    fill_f<<<1024, 256, 0, stream>>>(agg, zeroCount, 0.0f);
    fill_f<<<(N + 255) / 256, 256, 0, stream>>>(emax, (size_t)N, -3.402823466e38f);

    gemm_xlxr<<<(N / 16 + 7) / 8, 256, 0, stream>>>(x, packed, bl, br, xlp, xrp, N);

    edge_stats<<<(E + 255) / 256, 256, 0, stream>>>(dst, eattr, cnt, asum, E);
    loopattr_k<<<(N + 255) / 256, 256, 0, stream>>>(asum, cnt, N);

    edge_score<<<(ET + 7) / 8, 256, 0, stream>>>(src, dst, eattr, asum, xlp, xrp,
                                                 We, att, escore, emax, E, N);
    expdenom_k<<<(ET + 255) / 256, 256, 0, stream>>>(dst, escore, emax, denom, E, N);
    scatter_k<<<(ET + 7) / 8, 256, 0, stream>>>(src, dst, escore, denom, xlp, agg, E, N);

    pool_k<<<512, 256, 0, stream>>>(agg, b_conv, pooled, (size_t)N * C_CH);
    head_k<<<1, 64, 0, stream>>>(pooled, W_head, b_head, (float*)d_out, 1.0f / (float)N);
}
